// SelfAttention_12438225289818
// MI455X (gfx1250) — matrix operand order repaired and checked by reading, not compiled, not run on hardware
//
#include <hip/hip_runtime.h>

typedef __attribute__((ext_vector_type(16))) __bf16 v16bf;
typedef __attribute__((ext_vector_type(8)))  __bf16 v8bf;
typedef __attribute__((ext_vector_type(8)))  float  v8f;
typedef unsigned short u16;

#define SEQ   4096
#define DE    512
#define NH    8
#define DHD   64
#define BATCH 2
#define MTOT  (BATCH*SEQ)   // 8192 rows total

__device__ __forceinline__ u16 f2bf(float f) {
  __bf16 h = (__bf16)f;
  return __builtin_bit_cast(u16, h);
}

__device__ __forceinline__ v16bf join16(v8bf lo, v8bf hi) {
  v16bf a;
#pragma unroll
  for (int j = 0; j < 8; ++j) { a[j] = lo[j]; a[j + 8] = hi[j]; }
  return a;
}

__device__ __forceinline__ v8f wmma_bf16(v16bf a, v16bf b, v8f c) {
  return __builtin_amdgcn_wmma_f32_16x16x32_bf16(false, a, false, b, (short)0, c,
                                                 false, false);
}

// ---------------------------------------------------------------- fp32 -> bf16
__global__ void cvt_bf16_kernel(const float* __restrict__ src,
                                u16* __restrict__ dst, int n) {
  int i = blockIdx.x * blockDim.x + threadIdx.x;
  int stride = gridDim.x * blockDim.x;
  for (; i < n; i += stride) dst[i] = f2bf(src[i]);
}

// ------------------------------------------------- fused QKV GEMM + bias + SiLU
// q[m,e] = silu(sum_d x[m,d]*w[e,d] + b[e]); output in [b,h,s,dh] bf16 layout.
// One wave computes a 16x64 output tile; N dim fused over {Q,K,V} = 1536.
__global__ __launch_bounds__(256)
void qkv_gemm_kernel(const u16* __restrict__ X,
                     const u16* __restrict__ Wq, const u16* __restrict__ Wk,
                     const u16* __restrict__ Wv,
                     const float* __restrict__ bq, const float* __restrict__ bk,
                     const float* __restrict__ bv,
                     u16* __restrict__ Q, u16* __restrict__ K,
                     u16* __restrict__ V) {
  const int lane = threadIdx.x & 31;
  const int wave = threadIdx.x >> 5;
  const int w = blockIdx.x * 8 + wave;
  const int mt = w / 24, nt = w % 24;
  const int m0 = mt * 16;
  const int n0g = nt * 64;
  const int g = lane >> 4, r = lane & 15;
  const int mat = n0g >> 9;     // which of {Q,K,V}
  const int e0 = n0g & 511;
  const u16*   Wm = (mat == 0) ? Wq : (mat == 1) ? Wk : Wv;
  const float* bm = (mat == 0) ? bq : (mat == 1) ? bk : bv;
  u16*         Om = (mat == 0) ? Q  : (mat == 1) ? K  : V;

  v8f a0 = {}, a1 = {}, a2 = {}, a3 = {};
  const u16* xrow = X + (size_t)(m0 + r) * DE;
  const u16* w0 = Wm + (size_t)(e0 +      r) * DE;
  const u16* w1 = Wm + (size_t)(e0 + 16 + r) * DE;
  const u16* w2 = Wm + (size_t)(e0 + 32 + r) * DE;
  const u16* w3 = Wm + (size_t)(e0 + 48 + r) * DE;

  for (int kt = 0; kt < DE; kt += 32) {
    v16bf a = join16(*(const v8bf*)(xrow + kt + 16 * g),
                     *(const v8bf*)(xrow + kt + 16 * g + 8));
    v16bf b0 = join16(*(const v8bf*)(w0 + kt + 16 * g),
                      *(const v8bf*)(w0 + kt + 16 * g + 8));
    v16bf b1 = join16(*(const v8bf*)(w1 + kt + 16 * g),
                      *(const v8bf*)(w1 + kt + 16 * g + 8));
    v16bf b2 = join16(*(const v8bf*)(w2 + kt + 16 * g),
                      *(const v8bf*)(w2 + kt + 16 * g + 8));
    v16bf b3 = join16(*(const v8bf*)(w3 + kt + 16 * g),
                      *(const v8bf*)(w3 + kt + 16 * g + 8));
    a0 = wmma_bf16(a, b0, a0);
    a1 = wmma_bf16(a, b1, a1);
    a2 = wmma_bf16(a, b2, a2);
    a3 = wmma_bf16(a, b3, a3);
  }

#pragma unroll
  for (int f = 0; f < 4; ++f) {
    const int e = e0 + f * 16 + r;
    const float bias = bm[e];
    const int h = e >> 6, dh = e & 63;
    const v8f& acc = (f == 0) ? a0 : (f == 1) ? a1 : (f == 2) ? a2 : a3;
#pragma unroll
    for (int i = 0; i < 8; ++i) {
      const int m = m0 + i + 8 * g;
      float vx = acc[i] + bias;
      vx = vx / (1.0f + __expf(-vx));          // SiLU
      const int b = m >> 12;
      const int s = m & (SEQ - 1);
      Om[((size_t)(b * NH + h) * SEQ + s) * DHD + dh] = f2bf(vx);
    }
  }
}

// -------------------------------------------------------------- flash attention
// One wave owns 16 query rows; streams causal key tiles of 32 with online
// softmax.  S via QK^T WMMA; V tiles brought in with GLOBAL_LOAD_ASYNC_TO_LDS
// (ASYNCcnt, bypasses VGPRs, overlaps the S-WMMAs + softmax); P re-laid-out
// through LDS into an A-fragment.
__global__ __launch_bounds__(256)
void flash_attn_kernel(const u16* __restrict__ Q, const u16* __restrict__ K,
                       const u16* __restrict__ V, u16* __restrict__ Ctx,
                       const int* __restrict__ causal_p) {
  __shared__ __attribute__((aligned(16))) u16 smem[8 * 2560];  // 40 KB
  const int lane = threadIdx.x & 31;
  const int wave = threadIdx.x >> 5;
  const int g = lane >> 4, r = lane & 15;
  u16* pP = smem + wave * 2560;  // 16x32 P tile
  u16* pV = pP + 512;            // 32x64 V tile
  const int bh = blockIdx.y;
  const int q0 = (blockIdx.x * 8 + wave) * 16;
  const u16* Qb = Q + (size_t)bh * SEQ * DHD;
  const u16* Kb = K + (size_t)bh * SEQ * DHD;
  const u16* Vb = V + (size_t)bh * SEQ * DHD;
  const int causal = causal_p[0];

  // LDS byte address of this wave's V tile, spread one b128 per lane
  const unsigned pv_lds = (unsigned)(size_t)(void*)pV + (unsigned)(lane * 16);

  // Q fragments are loop-invariant (A covers full d_head=64 in two K=32 halves)
  const u16* qrow = Qb + (size_t)(q0 + r) * DHD;
  v16bf aQ0 = join16(*(const v8bf*)(qrow + 16 * g),
                     *(const v8bf*)(qrow + 16 * g + 8));
  v16bf aQ1 = join16(*(const v8bf*)(qrow + 32 + 16 * g),
                     *(const v8bf*)(qrow + 40 + 16 * g));

  v8f O0 = {}, O1 = {}, O2 = {}, O3 = {};
  float mrun[8], lsum[8];
#pragma unroll
  for (int i = 0; i < 8; ++i) { mrun[i] = -3.0e38f; lsum[i] = 0.0f; }

  const int kend = causal ? (q0 + 15) : (SEQ - 1);
  for (int k0 = 0; k0 <= kend; k0 += 32) {
    // ---- kick off async V tile copy (32x64 bf16, contiguous, 8 x 512B)
    // WAR-safe: previous iteration's ds_load gathers completed before their
    // consuming WMMAs (compiler-inserted s_wait_dscnt on the VGPR deps).
    {
      unsigned long long gsrc =
          (unsigned long long)(size_t)(Vb + (size_t)k0 * DHD) +
          (unsigned long long)(lane * 16);
      asm volatile(
          "global_load_async_to_lds_b128 %0, %1, off\n\t"
          "global_load_async_to_lds_b128 %0, %1, off offset:512\n\t"
          "global_load_async_to_lds_b128 %0, %1, off offset:1024\n\t"
          "global_load_async_to_lds_b128 %0, %1, off offset:1536\n\t"
          "global_load_async_to_lds_b128 %0, %1, off offset:2048\n\t"
          "global_load_async_to_lds_b128 %0, %1, off offset:2560\n\t"
          "global_load_async_to_lds_b128 %0, %1, off offset:3072\n\t"
          "global_load_async_to_lds_b128 %0, %1, off offset:3584"
          :: "v"(pv_lds), "v"(gsrc) : "memory");
    }

    // ---- S = Q * K^T for 32 keys (two 16-col fragments)
    v8f sc0 = {}, sc1 = {};
    {
      const u16* krow0 = Kb + (size_t)(k0 + r) * DHD;
      const u16* krow1 = Kb + (size_t)(k0 + 16 + r) * DHD;
      v16bf bK00 = join16(*(const v8bf*)(krow0 + 16 * g),
                          *(const v8bf*)(krow0 + 16 * g + 8));
      v16bf bK01 = join16(*(const v8bf*)(krow0 + 32 + 16 * g),
                          *(const v8bf*)(krow0 + 40 + 16 * g));
      v16bf bK10 = join16(*(const v8bf*)(krow1 + 16 * g),
                          *(const v8bf*)(krow1 + 16 * g + 8));
      v16bf bK11 = join16(*(const v8bf*)(krow1 + 32 + 16 * g),
                          *(const v8bf*)(krow1 + 40 + 16 * g));
      sc0 = wmma_bf16(aQ0, bK00, sc0);
      sc0 = wmma_bf16(aQ1, bK01, sc0);
      sc1 = wmma_bf16(aQ0, bK10, sc1);
      sc1 = wmma_bf16(aQ1, bK11, sc1);
    }

    // ---- online softmax per query row (row i+8g lives across lanes of group g)
#pragma unroll
    for (int i = 0; i < 8; ++i) {
      const int qg = q0 + i + 8 * g;
      float s0 = sc0[i] * 0.125f;   // 1/sqrt(64)
      float s1 = sc1[i] * 0.125f;
      if (causal) {
        if (k0 + r > qg)       s0 = -3.0e38f;
        if (k0 + 16 + r > qg)  s1 = -3.0e38f;
      }
      float mc = fmaxf(s0, s1);
      mc = fmaxf(mc, __shfl_xor(mc, 1, 32));
      mc = fmaxf(mc, __shfl_xor(mc, 2, 32));
      mc = fmaxf(mc, __shfl_xor(mc, 4, 32));
      mc = fmaxf(mc, __shfl_xor(mc, 8, 32));
      const float mn = fmaxf(mrun[i], mc);
      const float alpha = __expf(mrun[i] - mn);
      const float p0 = __expf(s0 - mn);
      const float p1 = __expf(s1 - mn);
      float rs = p0 + p1;
      rs += __shfl_xor(rs, 1, 32);
      rs += __shfl_xor(rs, 2, 32);
      rs += __shfl_xor(rs, 4, 32);
      rs += __shfl_xor(rs, 8, 32);
      lsum[i] = lsum[i] * alpha + rs;
      mrun[i] = mn;
      O0[i] *= alpha; O1[i] *= alpha; O2[i] *= alpha; O3[i] *= alpha;
      const int m = i + 8 * g;
      pP[m * 32 + r]      = f2bf(p0);
      pP[m * 32 + 16 + r] = f2bf(p1);
    }

    // P tile: cross-lane LDS reuse needs an explicit DS wait.
    asm volatile("s_wait_dscnt 0x0" ::: "memory");
    // V tile: async copy must have landed before the gathers below.
    asm volatile("s_wait_asynccnt 0x0" ::: "memory");

    // ---- O += P * V
    const u16* prow = pP + r * 32;
    v16bf aP = join16(*(const v8bf*)(prow + 16 * g),
                      *(const v8bf*)(prow + 16 * g + 8));
    v16bf bV0, bV1, bV2, bV3;
#pragma unroll
    for (int j = 0; j < 16; ++j) {
      const int rowoff = (16 * g + j) * 64 + r;
      bV0[j] = __builtin_bit_cast(__bf16, pV[rowoff + 0]);
      bV1[j] = __builtin_bit_cast(__bf16, pV[rowoff + 16]);
      bV2[j] = __builtin_bit_cast(__bf16, pV[rowoff + 32]);
      bV3[j] = __builtin_bit_cast(__bf16, pV[rowoff + 48]);
    }
    O0 = wmma_bf16(aP, bV0, O0);
    O1 = wmma_bf16(aP, bV1, O1);
    O2 = wmma_bf16(aP, bV2, O2);
    O3 = wmma_bf16(aP, bV3, O3);
  }

  // ---- normalize and write context in [b,s,(h,dh)] bf16 layout
  const int b = bh >> 3, h = bh & 7;
#pragma unroll
  for (int i = 0; i < 8; ++i) {
    const int m = i + 8 * g;
    const size_t row = (size_t)(b * SEQ + q0 + m) * DE + h * DHD;
    const float inv = 1.0f / lsum[i];
    Ctx[row + r]      = f2bf(O0[i] * inv);
    Ctx[row + 16 + r] = f2bf(O1[i] * inv);
    Ctx[row + 32 + r] = f2bf(O2[i] * inv);
    Ctx[row + 48 + r] = f2bf(O3[i] * inv);
  }
}

// ----------------------------------------------------- output projection + bias
__global__ __launch_bounds__(256)
void out_gemm_kernel(const u16* __restrict__ C, const u16* __restrict__ Wo,
                     const float* __restrict__ bo, float* __restrict__ Out) {
  const int lane = threadIdx.x & 31;
  const int wave = threadIdx.x >> 5;
  const int w = blockIdx.x * 8 + wave;
  const int mt = w >> 3, nt = w & 7;
  const int m0 = mt * 16, n0 = nt * 64;
  const int g = lane >> 4, r = lane & 15;

  v8f a0 = {}, a1 = {}, a2 = {}, a3 = {};
  const u16* xrow = C + (size_t)(m0 + r) * DE;
  const u16* w0 = Wo + (size_t)(n0 +      r) * DE;
  const u16* w1 = Wo + (size_t)(n0 + 16 + r) * DE;
  const u16* w2 = Wo + (size_t)(n0 + 32 + r) * DE;
  const u16* w3 = Wo + (size_t)(n0 + 48 + r) * DE;

  for (int kt = 0; kt < DE; kt += 32) {
    v16bf a = join16(*(const v8bf*)(xrow + kt + 16 * g),
                     *(const v8bf*)(xrow + kt + 16 * g + 8));
    v16bf b0 = join16(*(const v8bf*)(w0 + kt + 16 * g),
                      *(const v8bf*)(w0 + kt + 16 * g + 8));
    v16bf b1 = join16(*(const v8bf*)(w1 + kt + 16 * g),
                      *(const v8bf*)(w1 + kt + 16 * g + 8));
    v16bf b2 = join16(*(const v8bf*)(w2 + kt + 16 * g),
                      *(const v8bf*)(w2 + kt + 16 * g + 8));
    v16bf b3 = join16(*(const v8bf*)(w3 + kt + 16 * g),
                      *(const v8bf*)(w3 + kt + 16 * g + 8));
    a0 = wmma_bf16(a, b0, a0);
    a1 = wmma_bf16(a, b1, a1);
    a2 = wmma_bf16(a, b2, a2);
    a3 = wmma_bf16(a, b3, a3);
  }

#pragma unroll
  for (int f = 0; f < 4; ++f) {
    const int n = n0 + f * 16 + r;
    const float bias = bo[n];
    const v8f& acc = (f == 0) ? a0 : (f == 1) ? a1 : (f == 2) ? a2 : a3;
#pragma unroll
    for (int i = 0; i < 8; ++i) {
      const int m = m0 + i + 8 * g;
      Out[(size_t)m * DE + n] = acc[i] + bias;
    }
  }
}

// ------------------------------------------------------------------------ host
extern "C" void kernel_launch(void* const* d_in, const int* in_sizes, int n_in,
                              void* d_out, int out_size, void* d_ws,
                              size_t ws_size, hipStream_t stream) {
  const float* x  = (const float*)d_in[0];
  const float* wq = (const float*)d_in[1];
  const float* bq = (const float*)d_in[2];
  const float* wk = (const float*)d_in[3];
  const float* bk = (const float*)d_in[4];
  const float* wv = (const float*)d_in[5];
  const float* bv = (const float*)d_in[6];
  const float* wo = (const float*)d_in[7];
  const float* bo = (const float*)d_in[8];
  const int* causal = (const int*)d_in[9];

  u16* ws  = (u16*)d_ws;
  u16* xb  = ws;                            // 8192*512 bf16
  u16* wqb = xb  + (size_t)MTOT * DE;
  u16* wkb = wqb + (size_t)DE * DE;
  u16* wvb = wkb + (size_t)DE * DE;
  u16* wob = wvb + (size_t)DE * DE;
  u16* Qw  = wob + (size_t)DE * DE;         // [b,h,s,dh]
  u16* Kw  = Qw  + (size_t)MTOT * DE;
  u16* Vw  = Kw  + (size_t)MTOT * DE;
  u16* Cw  = Vw  + (size_t)MTOT * DE;       // context [m, e]

  cvt_bf16_kernel<<<256, 256, 0, stream>>>(x,  xb,  MTOT * DE);
  cvt_bf16_kernel<<<64,  256, 0, stream>>>(wq, wqb, DE * DE);
  cvt_bf16_kernel<<<64,  256, 0, stream>>>(wk, wkb, DE * DE);
  cvt_bf16_kernel<<<64,  256, 0, stream>>>(wv, wvb, DE * DE);
  cvt_bf16_kernel<<<64,  256, 0, stream>>>(wo, wob, DE * DE);

  // 512 M-tiles x 24 fused-N tiles, 8 waves/block
  qkv_gemm_kernel<<<1536, 256, 0, stream>>>(xb, wqb, wkb, wvb, bq, bk, bv,
                                            Qw, Kw, Vw);
  // 32 q-tile groups (8 waves of 16 rows each) x 16 (b,h) pairs
  flash_attn_kernel<<<dim3(32, 16), 256, 0, stream>>>(Qw, Kw, Vw, Cw, causal);
  // 512 M-tiles x 8 N-tiles
  out_gemm_kernel<<<512, 256, 0, stream>>>(Cw, wob, bo, (float*)d_out);
}
